// TransformerConvLayer_14078902796420
// MI455X (gfx1250) — compile-verified, matrix operand order checked by reading, not run on hardware
//
#include <hip/hip_runtime.h>
#include <hip/hip_bf16.h>
#include <stdint.h>

// Problem constants (fixed by the reference).
#define N_NODES 8192
#define IN_DIM  512
#define OUT_DIM 512
#define NHEAD   4
#define HC      2048           // NHEAD * OUT_DIM
#define NEDGE   131072

typedef __attribute__((ext_vector_type(16))) __bf16 v16bf;
typedef __attribute__((ext_vector_type(8)))  float  v8f;

union FragAB { v16bf v; uint32_t u[8]; uint16_t s[16]; };
union FragC  { v8f   v; float    f[8]; };

__device__ __forceinline__ uint16_t f2bf(float x) {
  uint32_t u = __float_as_uint(x);
  u += 0x7fffu + ((u >> 16) & 1u);        // round-to-nearest-even
  return (uint16_t)(u >> 16);
}

// ---------------------------------------------------------------------------
// bf16 WMMA GEMM:  C[M x Nw] = A[M x K] @ W[K x Nw] + bias (+ resid)
// Block tile 128(M) x 256(N), K-step 32. 256 threads = 8 waves (2 M x 4 N),
// each wave computes 64x64 = 4x4 WMMA 16x16x32 tiles (16 WMMA / K-step).
// Double-buffered LDS + register prefetch of the next K-tile.
// ---------------------------------------------------------------------------
#define BM 128
#define BN 256
#define BK 32
#define LDSS 40   // padded row stride in bf16 elems (80B: 16B-aligned, bank-spread)

__global__ __launch_bounds__(256)
void gemm_bf16_kernel(const float* __restrict__ A, const float* __restrict__ W,
                      const float* __restrict__ bias, const float* __restrict__ resid,
                      int resid_ld, float* __restrict__ C, int K, int Nw)
{
  __shared__ uint16_t sA[2][BM * LDSS];
  __shared__ uint16_t sB[2][BN * LDSS];   // W tile stored transposed: sB[n][k]

  const int tid   = threadIdx.x;
  const int lane  = tid & 31;
  const int wave  = tid >> 5;
  const int waveM = wave >> 2;                 // 0..1 -> 64-row span
  const int waveN = wave & 3;                  // 0..3 -> 64-col span
  const int rowBase = blockIdx.y * BM;
  const int colBase = blockIdx.x * BN;
  const int half = lane >> 4;                  // K-group select
  const int mr   = lane & 15;

  // Prefetch staging registers for the next K-tile.
  float4 ra[4];   // A: 128x32 = 1024 float4 / 256 thr
  float4 rb[8];   // W: 32x256 = 2048 float4 / 256 thr

  FragC acc[4][4];
#pragma unroll
  for (int mt = 0; mt < 4; ++mt)
#pragma unroll
    for (int nt = 0; nt < 4; ++nt)
#pragma unroll
      for (int r = 0; r < 8; ++r) acc[mt][nt].f[r] = 0.f;

  auto loadA = [&](int k0) {
#pragma unroll
    for (int i = 0; i < 4; ++i) {
      int f  = tid + i * 256;
      int r  = f >> 3;
      int kq = (f & 7) << 2;
      ra[i] = *(const float4*)(A + (size_t)(rowBase + r) * K + k0 + kq);
    }
  };
  auto loadB = [&](int k0) {
#pragma unroll
    for (int i = 0; i < 8; ++i) {
      int f  = tid + i * 256;
      int kk = f >> 6;               // 0..31
      int nq = (f & 63) << 2;        // 0..252
      rb[i] = *(const float4*)(W + (size_t)(k0 + kk) * Nw + colBase + nq);
    }
  };
  auto stageA = [&](int buf) {
#pragma unroll
    for (int i = 0; i < 4; ++i) {
      int f  = tid + i * 256;
      int r  = f >> 3;
      int kq = (f & 7) << 2;
      uint32_t p0 = (uint32_t)f2bf(ra[i].x) | ((uint32_t)f2bf(ra[i].y) << 16);
      uint32_t p1 = (uint32_t)f2bf(ra[i].z) | ((uint32_t)f2bf(ra[i].w) << 16);
      uint16_t* d = &sA[buf][r * LDSS + kq];
      *(uint32_t*)(d + 0) = p0;
      *(uint32_t*)(d + 2) = p1;
    }
  };
  auto stageB = [&](int buf) {
#pragma unroll
    for (int i = 0; i < 8; ++i) {
      int f  = tid + i * 256;
      int kk = f >> 6;
      int nq = (f & 63) << 2;
      sB[buf][(nq + 0) * LDSS + kk] = f2bf(rb[i].x);
      sB[buf][(nq + 1) * LDSS + kk] = f2bf(rb[i].y);
      sB[buf][(nq + 2) * LDSS + kk] = f2bf(rb[i].z);
      sB[buf][(nq + 3) * LDSS + kk] = f2bf(rb[i].w);
    }
  };

  // Prologue: stage first K-tile.
  loadA(0); loadB(0);
  stageA(0); stageB(0);
  __syncthreads();

  int buf = 0;
  for (int k0 = 0; k0 < K; k0 += BK) {
    const bool more = (k0 + BK) < K;
    if (more) { loadA(k0 + BK); loadB(k0 + BK); }   // global loads in flight during WMMA

    // B fragments (ISA layout; sB is column-major so N plays the role of M).
    FragAB fb[4];
#pragma unroll
    for (int nt = 0; nt < 4; ++nt) {
      int nrow = waveN * 64 + nt * 16 + mr;
#pragma unroll
      for (int v = 0; v < 8; ++v) {
        int kb = ((v >> 2) << 4) + (half << 3) + ((v & 3) << 1);
        fb[nt].u[v] = *(const uint32_t*)&sB[buf][nrow * LDSS + kb];
      }
    }
#pragma unroll
    for (int mt = 0; mt < 4; ++mt) {
      FragAB fa;
      int mrow = waveM * 64 + mt * 16 + mr;
#pragma unroll
      for (int v = 0; v < 8; ++v) {
        int kb = ((v >> 2) << 4) + (half << 3) + ((v & 3) << 1);
        fa.u[v] = *(const uint32_t*)&sA[buf][mrow * LDSS + kb];
      }
#pragma unroll
      for (int nt = 0; nt < 4; ++nt) {
        acc[mt][nt].v = __builtin_amdgcn_wmma_f32_16x16x32_bf16(
            false, fa.v, false, fb[nt].v, (short)0, acc[mt][nt].v, false, false);
      }
    }

    if (more) {
      stageA(buf ^ 1); stageB(buf ^ 1);   // write other buffer: no barrier needed first
      buf ^= 1;
      __syncthreads();
    }
  }

  // Epilogue: D layout = VGPR r -> M = r + 8*half, N = lane&15.
  const bool hasRes = (resid != nullptr);
  const int nc = lane & 15;
#pragma unroll
  for (int mt = 0; mt < 4; ++mt) {
#pragma unroll
    for (int nt = 0; nt < 4; ++nt) {
      int col = colBase + waveN * 64 + nt * 16 + nc;
      float b = bias[col];
#pragma unroll
      for (int r = 0; r < 8; ++r) {
        int row = rowBase + waveM * 64 + mt * 16 + r + half * 8;
        float val = acc[mt][nt].f[r] + b;
        if (hasRes) val += resid[(size_t)row * resid_ld + col];
        C[(size_t)row * Nw + col] = val;
      }
    }
  }
}

// ---------------------------------------------------------------------------
// CSR build over destination nodes
// ---------------------------------------------------------------------------
__global__ void zero2_kernel(int* __restrict__ a, int* __restrict__ b, int n) {
  int i = blockIdx.x * blockDim.x + threadIdx.x;
  if (i < n) { a[i] = 0; b[i] = 0; }
}

__global__ void count_kernel(const long long* __restrict__ ei, int* __restrict__ counts) {
  int e = blockIdx.x * blockDim.x + threadIdx.x;
  if (e < NEDGE) atomicAdd(&counts[(int)ei[NEDGE + e]], 1);
}

__global__ __launch_bounds__(256)
void scan_kernel(const int* __restrict__ counts, int* __restrict__ rowptr) {
  __shared__ int part[256];
  const int t = threadIdx.x;
  const int base = t * 32;
  int local[32];
  int s = 0;
#pragma unroll
  for (int i = 0; i < 32; ++i) { local[i] = s; s += counts[base + i]; }
  part[t] = s;
  __syncthreads();
  for (int off = 1; off < 256; off <<= 1) {
    int v = (t >= off) ? part[t - off] : 0;
    __syncthreads();
    part[t] += v;
    __syncthreads();
  }
  int pre = (t == 0) ? 0 : part[t - 1];
#pragma unroll
  for (int i = 0; i < 32; ++i) rowptr[base + i] = pre + local[i];
  if (t == 255) rowptr[N_NODES] = part[255];
}

__global__ void fill_kernel(const long long* __restrict__ ei, const int* __restrict__ rowptr,
                            int* __restrict__ cursor, int* __restrict__ es_src,
                            int* __restrict__ es_dst) {
  int e = blockIdx.x * blockDim.x + threadIdx.x;
  if (e >= NEDGE) return;
  int s = (int)ei[e];
  int d = (int)ei[NEDGE + e];
  int pos = rowptr[d] + atomicAdd(&cursor[d], 1);
  es_src[pos] = s;
  es_dst[pos] = d;
}

// ---------------------------------------------------------------------------
// Per-edge attention logits: one wave per edge, alpha[e,h] = q[d,h]·k[s,h]/sqrt(C)
// ---------------------------------------------------------------------------
__global__ __launch_bounds__(256)
void alpha_kernel(const float* __restrict__ q, const float* __restrict__ k,
                  const int* __restrict__ es_src, const int* __restrict__ es_dst,
                  float* __restrict__ alphas) {
  int gw   = (blockIdx.x * 256 + threadIdx.x) >> 5;
  int lane = threadIdx.x & 31;
  if (gw >= NEDGE) return;
  int s = es_src[gw], d = es_dst[gw];
  const float4* qp = (const float4*)(q + (size_t)d * HC);
  const float4* kp = (const float4*)(k + (size_t)s * HC);
#pragma unroll
  for (int h = 0; h < NHEAD; ++h) {
    float acc = 0.f;
#pragma unroll
    for (int i = 0; i < 4; ++i) {
      int f = h * 128 + lane + i * 32;
      float4 a = qp[f], b = kp[f];
      acc += a.x * b.x + a.y * b.y + a.z * b.z + a.w * b.w;
    }
#pragma unroll
    for (int off = 16; off; off >>= 1) acc += __shfl_xor(acc, off, 32);
    if (lane == 0) alphas[(size_t)gw * NHEAD + h] = acc * 0.04419417382415922f; // 1/sqrt(512)
  }
}

// ---------------------------------------------------------------------------
// Per-node segment softmax + weighted gather of v (adds onto skip already in xattn)
// ---------------------------------------------------------------------------
__global__ __launch_bounds__(256)
void agg_kernel(const float* __restrict__ vbuf, const float* __restrict__ alphas,
                const int* __restrict__ es_src, const int* __restrict__ rowptr,
                float* __restrict__ xattn) {
  __shared__ float mx[NHEAD], rden[NHEAD];
  const int n = blockIdx.x, t = threadIdx.x;
  const int beg = rowptr[n], end = rowptr[n + 1];
  if (t < NHEAD) {
    float m = -3.4e38f;
    for (int i = beg; i < end; ++i) m = fmaxf(m, alphas[(size_t)i * NHEAD + t]);
    float sum = 0.f;
    for (int i = beg; i < end; ++i) sum += __expf(alphas[(size_t)i * NHEAD + t] - m);
    mx[t] = m;
    rden[t] = (sum > 0.f) ? 1.f / sum : 0.f;
  }
  __syncthreads();
  const int c0 = t * 8;            // 8 cols per thread, single head each
  const int head = t >> 6;
  float4 a0 = *(float4*)(xattn + (size_t)n * HC + c0);
  float4 a1 = *(float4*)(xattn + (size_t)n * HC + c0 + 4);
  const float m = mx[head], rd = rden[head];
  for (int i = beg; i < end; ++i) {
    int s = es_src[i];
    float w = __expf(alphas[(size_t)i * NHEAD + head] - m) * rd;
    const float4* vp = (const float4*)(vbuf + (size_t)s * HC + c0);
    float4 v0 = vp[0], v1 = vp[1];
    a0.x += w * v0.x; a0.y += w * v0.y; a0.z += w * v0.z; a0.w += w * v0.w;
    a1.x += w * v1.x; a1.y += w * v1.y; a1.z += w * v1.z; a1.w += w * v1.w;
  }
  *(float4*)(xattn + (size_t)n * HC + c0)     = a0;
  *(float4*)(xattn + (size_t)n * HC + c0 + 4) = a1;
}

// ---------------------------------------------------------------------------
// Row LayerNorm over 512 channels
// ---------------------------------------------------------------------------
__global__ __launch_bounds__(256)
void ln_kernel(const float* __restrict__ h, const float* __restrict__ gamma,
               const float* __restrict__ beta, float* __restrict__ out) {
  __shared__ float red[256];
  const int n = blockIdx.x, t = threadIdx.x;
  const float* hp = h + (size_t)n * OUT_DIM;
  float a = hp[t], b = hp[t + 256];
  red[t] = a + b;
  __syncthreads();
  for (int off = 128; off; off >>= 1) { if (t < off) red[t] += red[t + off]; __syncthreads(); }
  float mu = red[0] * (1.f / OUT_DIM);
  __syncthreads();
  float da = a - mu, db = b - mu;
  red[t] = da * da + db * db;
  __syncthreads();
  for (int off = 128; off; off >>= 1) { if (t < off) red[t] += red[t + off]; __syncthreads(); }
  float rstd = rsqrtf(red[0] * (1.f / OUT_DIM) + 1e-5f);
  out[(size_t)n * OUT_DIM + t]       = gamma[t] * da * rstd + beta[t];
  out[(size_t)n * OUT_DIM + t + 256] = gamma[t + 256] * db * rstd + beta[t + 256];
}

// ---------------------------------------------------------------------------
extern "C" void kernel_launch(void* const* d_in, const int* in_sizes, int n_in,
                              void* d_out, int out_size, void* d_ws, size_t ws_size,
                              hipStream_t stream) {
  const float*     x     = (const float*)d_in[0];
  const long long* ei    = (const long long*)d_in[1];
  const float*     Wq    = (const float*)d_in[2];
  const float*     bq    = (const float*)d_in[3];
  const float*     Wk    = (const float*)d_in[4];
  const float*     bk    = (const float*)d_in[5];
  const float*     Wv    = (const float*)d_in[6];
  const float*     bv    = (const float*)d_in[7];
  const float*     Wsk   = (const float*)d_in[8];
  const float*     bsk   = (const float*)d_in[9];
  const float*     Wlin  = (const float*)d_in[10];
  const float*     blin  = (const float*)d_in[11];
  const float*     gamma = (const float*)d_in[12];
  const float*     beta  = (const float*)d_in[13];
  float* out = (float*)d_out;

  // Workspace carve-up
  char* p = (char*)d_ws;
  const size_t nodeMat = (size_t)N_NODES * HC * sizeof(float);      // 64 MB
  float* q     = (float*)p; p += nodeMat;
  float* kbuf  = (float*)p; p += nodeMat;
  float* vbuf  = (float*)p; p += nodeMat;
  float* xattn = (float*)p; p += nodeMat;
  float* hbuf  = (float*)p; p += (size_t)N_NODES * OUT_DIM * sizeof(float);
  float* alphas= (float*)p; p += (size_t)NEDGE * NHEAD * sizeof(float);
  int* counts  = (int*)p;   p += N_NODES * sizeof(int);
  int* cursor  = (int*)p;   p += N_NODES * sizeof(int);
  int* rowptr  = (int*)p;   p += (N_NODES + 1) * sizeof(int);
  int* es_src  = (int*)p;   p += NEDGE * sizeof(int);
  int* es_dst  = (int*)p;   p += NEDGE * sizeof(int);

  const dim3 blk(256);
  // Projections: q, k, v; skip goes straight into the aggregation buffer.
  const dim3 g1(HC / BN, N_NODES / BM);       // 8 x 64
  gemm_bf16_kernel<<<g1, blk, 0, stream>>>(x, Wq,  bq,  nullptr, 0, q,     IN_DIM, HC);
  gemm_bf16_kernel<<<g1, blk, 0, stream>>>(x, Wk,  bk,  nullptr, 0, kbuf,  IN_DIM, HC);
  gemm_bf16_kernel<<<g1, blk, 0, stream>>>(x, Wv,  bv,  nullptr, 0, vbuf,  IN_DIM, HC);
  gemm_bf16_kernel<<<g1, blk, 0, stream>>>(x, Wsk, bsk, nullptr, 0, xattn, IN_DIM, HC);

  // CSR build
  zero2_kernel<<<(N_NODES + 255) / 256, blk, 0, stream>>>(counts, cursor, N_NODES);
  count_kernel<<<NEDGE / 256, blk, 0, stream>>>(ei, counts);
  scan_kernel<<<1, blk, 0, stream>>>(counts, rowptr);
  fill_kernel<<<NEDGE / 256, blk, 0, stream>>>(ei, rowptr, cursor, es_src, es_dst);

  // Edge logits (one wave per edge), then per-node softmax + gather.
  alpha_kernel<<<NEDGE / 8, blk, 0, stream>>>(q, kbuf, es_src, es_dst, alphas);
  agg_kernel<<<N_NODES, blk, 0, stream>>>(vbuf, alphas, es_src, rowptr, xattn);

  // Output projection with fused residual (+ x_attn[:, :512]), then LayerNorm.
  const dim3 g2(OUT_DIM / BN, N_NODES / BM);  // 2 x 64
  gemm_bf16_kernel<<<g2, blk, 0, stream>>>(xattn, Wlin, blin, xattn, HC, hbuf, HC, OUT_DIM);
  ln_kernel<<<N_NODES, blk, 0, stream>>>(hbuf, gamma, beta, out);
}
